// MarlinAttention_30915174596899
// MI455X (gfx1250) — matrix operand order repaired and checked by reading, not compiled, not run on hardware
//
#include <hip/hip_runtime.h>

typedef __attribute__((ext_vector_type(16))) _Float16 v16h;
typedef __attribute__((ext_vector_type(8)))  _Float16 v8h;
typedef __attribute__((ext_vector_type(8)))  float    v8f;

#define SEQ   4096
#define DH    64
#define NH    8
#define BQ    64
#define BK    64
#define WBLK  16   // window (1024) / BK

#define WMMA16(A, B, C) \
    __builtin_amdgcn_wmma_f32_16x16x32_f16(false, (A), false, (B), (short)0, (C), false, false)

// Intra-wave LDS turnaround: LDS ops from one wave are hardware in-order; we
// only need to stop the compiler from reordering around the spill/reload.
#define WAVE_FENCE() do { asm volatile("" ::: "memory"); \
                          __builtin_amdgcn_wave_barrier(); \
                          asm volatile("" ::: "memory"); } while (0)

__global__ __launch_bounds__(128)
void swa_attn_fwd(const float* __restrict__ Q, const float* __restrict__ K,
                  const float* __restrict__ V, float* __restrict__ O)
{
    const int qb   = blockIdx.x;          // q block (0..63)
    const int head = blockIdx.y;          // head (0..7)
    const int tid  = threadIdx.x;
    const int wave = tid >> 5;            // 0..3
    const int lane = tid & 31;
    const int lmod = lane & 15;
    const int g    = lane >> 4;           // half-wave group

    __shared__ __align__(32) _Float16 sK [2][BK][80];   // K tiles [key][d], f16
    __shared__ __align__(32) _Float16 sVT[2][DH][80];   // V tiles transposed [d][key], f16
    __shared__ __align__(16) float    sS [4][16][68];   // per-wave raw scores [row][key]

    const size_t hoff = (size_t)head * SEQ * DH;
    const float* Qh = Q + hoff;
    const float* Kh = K + hoff;
    const float* Vh = V + hoff;
    float*       Oh = O + hoff;

    const float scale = 0.125f;           // 1/sqrt(64)

    // ---- Q tile -> WMMA A-layout f16 (pre-scaled). A 16x32: lane M = lmod,
    // element h -> Kdim(d) = h + g*16.
    const int qrow_tile = qb * BQ + wave * 16;
    const int rowg      = qrow_tile + lmod;      // this lane's row (row-layout phases)
    v16h aq0, aq1;                               // d 0..31 / 32..63
    {
        const float* qp = Qh + (size_t)rowg * DH;
        const float4 t0 = *(const float4*)(qp + g * 16);
        const float4 t1 = *(const float4*)(qp + g * 16 + 4);
        const float4 t2 = *(const float4*)(qp + g * 16 + 8);
        const float4 t3 = *(const float4*)(qp + g * 16 + 12);
        const float4 t4 = *(const float4*)(qp + 32 + g * 16);
        const float4 t5 = *(const float4*)(qp + 32 + g * 16 + 4);
        const float4 t6 = *(const float4*)(qp + 32 + g * 16 + 8);
        const float4 t7 = *(const float4*)(qp + 32 + g * 16 + 12);
        const float f0[16] = {t0.x,t0.y,t0.z,t0.w, t1.x,t1.y,t1.z,t1.w,
                              t2.x,t2.y,t2.z,t2.w, t3.x,t3.y,t3.z,t3.w};
        const float f1[16] = {t4.x,t4.y,t4.z,t4.w, t5.x,t5.y,t5.z,t5.w,
                              t6.x,t6.y,t6.z,t6.w, t7.x,t7.y,t7.z,t7.w};
        #pragma unroll
        for (int h2 = 0; h2 < 16; ++h2) {
            aq0[h2] = (_Float16)(f0[h2] * scale);
            aq1[h2] = (_Float16)(f1[h2] * scale);
        }
    }

    // ---- Running state: O accumulators in C-layout (rows j + g*8), softmax
    // stats in row-layout (one row per lane: row = lmod).
    v8f o0 = {}, o1 = {}, o2 = {}, o3 = {};
    float mrow = -INFINITY, lrow = 0.0f;

    int jb0 = qb - (WBLK - 1); if (jb0 < 0) jb0 = 0;
    const int nblk = qb - jb0 + 1;

    // ---- Staging ownership: each thread owns half a key-row of the tile.
    const int srow = tid >> 1;            // row in k-block
    const int shh  = (tid & 1) * 32;      // first of 32 d-columns
    const float* kp = Kh + (size_t)(jb0 * BK + srow) * DH + shh;
    const float* vp = Vh + (size_t)(jb0 * BK + srow) * DH + shh;

    // Raw float4 staging registers (held across compute for latency overlap).
    float4 kr[8], vr[8];

    // ---- Preload block jb0 into buffer 0.
    #pragma unroll
    for (int i = 0; i < 8; ++i) { kr[i] = ((const float4*)kp)[i]; vr[i] = ((const float4*)vp)[i]; }
    #pragma unroll
    for (int i = 0; i < 4; ++i) {
        const float4 k0 = kr[2*i], k1 = kr[2*i+1];
        v8h hk;
        hk[0]=(_Float16)k0.x; hk[1]=(_Float16)k0.y; hk[2]=(_Float16)k0.z; hk[3]=(_Float16)k0.w;
        hk[4]=(_Float16)k1.x; hk[5]=(_Float16)k1.y; hk[6]=(_Float16)k1.z; hk[7]=(_Float16)k1.w;
        *(v8h*)&sK[0][srow][shh + i * 8] = hk;
        const float4 v0 = vr[2*i], v1 = vr[2*i+1];
        sVT[0][shh + i*8 + 0][srow] = (_Float16)v0.x;
        sVT[0][shh + i*8 + 1][srow] = (_Float16)v0.y;
        sVT[0][shh + i*8 + 2][srow] = (_Float16)v0.z;
        sVT[0][shh + i*8 + 3][srow] = (_Float16)v0.w;
        sVT[0][shh + i*8 + 4][srow] = (_Float16)v1.x;
        sVT[0][shh + i*8 + 5][srow] = (_Float16)v1.y;
        sVT[0][shh + i*8 + 6][srow] = (_Float16)v1.z;
        sVT[0][shh + i*8 + 7][srow] = (_Float16)v1.w;
    }
    kp += BK * DH; vp += BK * DH;
    __syncthreads();

    int cur = 0;
    for (int b = 0; b < nblk; ++b) {
        const int  jb        = jb0 + b;
        const bool have_next = (b + 1 < nblk);

        // Prefetch block jb+2 (lowers to global_prefetch_b8).
        if (b + 2 < nblk) {
            __builtin_prefetch(kp + BK * DH, 0, 3);
            __builtin_prefetch(vp + BK * DH, 0, 3);
        }
        // Issue next block's global loads; they overlap this block's compute.
        if (have_next) {
            #pragma unroll
            for (int i = 0; i < 8; ++i) { kr[i] = ((const float4*)kp)[i]; vr[i] = ((const float4*)vp)[i]; }
            kp += BK * DH; vp += BK * DH;
        }

        // ---- S = Q * K^T over all 64 keys: 4 C-tiles x 2 d-chunks = 8 WMMAs.
        // B 32x16: lane N = lmod (key), element h -> Kdim(d) = h + g*16.
        v8f s0 = {}, s1 = {}, s2 = {}, s3 = {};
        {
            v16h bmat;
            bmat = *(const v16h*)&sK[cur][ 0 + lmod][     g * 16]; s0 = WMMA16(aq0, bmat, s0);
            bmat = *(const v16h*)&sK[cur][ 0 + lmod][32 + g * 16]; s0 = WMMA16(aq1, bmat, s0);
            bmat = *(const v16h*)&sK[cur][16 + lmod][     g * 16]; s1 = WMMA16(aq0, bmat, s1);
            bmat = *(const v16h*)&sK[cur][16 + lmod][32 + g * 16]; s1 = WMMA16(aq1, bmat, s1);
            bmat = *(const v16h*)&sK[cur][32 + lmod][     g * 16]; s2 = WMMA16(aq0, bmat, s2);
            bmat = *(const v16h*)&sK[cur][32 + lmod][32 + g * 16]; s2 = WMMA16(aq1, bmat, s2);
            bmat = *(const v16h*)&sK[cur][48 + lmod][     g * 16]; s3 = WMMA16(aq0, bmat, s3);
            bmat = *(const v16h*)&sK[cur][48 + lmod][32 + g * 16]; s3 = WMMA16(aq1, bmat, s3);
        }

        // ---- Spill raw scores (C-layout) to this wave's LDS slice.
        #pragma unroll
        for (int j = 0; j < 8; ++j) {
            sS[wave][j + g * 8][ 0 + lmod] = s0[j];
            sS[wave][j + g * 8][16 + lmod] = s1[j];
            sS[wave][j + g * 8][32 + lmod] = s2[j];
            sS[wave][j + g * 8][48 + lmod] = s3[j];
        }
        WAVE_FENCE();   // intra-wave turnaround; LDS is in-order per wave

        // ---- Reload one full row per lane, directly at A-layout positions:
        // a0[h] = S[row=lmod][key = h + g*16], a1 = same + 32.
        float a0[16], a1[16];
        {
            const float* rp = &sS[wave][lmod][0];
            const float4 t0 = *(const float4*)(rp + g * 16);
            const float4 t1 = *(const float4*)(rp + g * 16 + 4);
            const float4 t2 = *(const float4*)(rp + g * 16 + 8);
            const float4 t3 = *(const float4*)(rp + g * 16 + 12);
            const float4 t4 = *(const float4*)(rp + 32 + g * 16);
            const float4 t5 = *(const float4*)(rp + 32 + g * 16 + 4);
            const float4 t6 = *(const float4*)(rp + 32 + g * 16 + 8);
            const float4 t7 = *(const float4*)(rp + 32 + g * 16 + 12);
            a0[ 0]=t0.x; a0[ 1]=t0.y; a0[ 2]=t0.z; a0[ 3]=t0.w;
            a0[ 4]=t1.x; a0[ 5]=t1.y; a0[ 6]=t1.z; a0[ 7]=t1.w;
            a0[ 8]=t2.x; a0[ 9]=t2.y; a0[10]=t2.z; a0[11]=t2.w;
            a0[12]=t3.x; a0[13]=t3.y; a0[14]=t3.z; a0[15]=t3.w;
            a1[ 0]=t4.x; a1[ 1]=t4.y; a1[ 2]=t4.z; a1[ 3]=t4.w;
            a1[ 4]=t5.x; a1[ 5]=t5.y; a1[ 6]=t5.z; a1[ 7]=t5.w;
            a1[ 8]=t6.x; a1[ 9]=t6.y; a1[10]=t6.z; a1[11]=t6.w;
            a1[12]=t7.x; a1[13]=t7.y; a1[14]=t7.z; a1[15]=t7.w;
        }

        // ---- Causal mask (diagonal block only).
        if (jb == qb) {
            #pragma unroll
            for (int h2 = 0; h2 < 16; ++h2) {
                const int c  = h2 + g * 16;
                const int kg = jb * BK + c;
                if (kg      > rowg) a0[h2] = -INFINITY;
                if (kg + 32 > rowg) a1[h2] = -INFINITY;
            }
        }

        // ---- Online softmax: local trees + one cross-half combine each.
        float mloc = a0[0];
        #pragma unroll
        for (int h2 = 1; h2 < 16; ++h2) mloc = fmaxf(mloc, a0[h2]);
        #pragma unroll
        for (int h2 = 0; h2 < 16; ++h2) mloc = fmaxf(mloc, a1[h2]);
        mloc = fmaxf(mloc, __shfl_xor(mloc, 16, 32));
        const float mnew  = fmaxf(mrow, mloc);
        const float alpha = (mrow == -INFINITY) ? 0.0f : __expf(mrow - mnew);
        float rs = 0.0f;
        v16h ap0, ap1;
        #pragma unroll
        for (int h2 = 0; h2 < 16; ++h2) {
            const float p0 = __expf(a0[h2] - mnew);   // -inf -> 0
            const float p1 = __expf(a1[h2] - mnew);
            rs += p0 + p1;
            ap0[h2] = (_Float16)p0;
            ap1[h2] = (_Float16)p1;
        }
        rs += __shfl_xor(rs, 16, 32);
        lrow = lrow * alpha + rs;
        mrow = mnew;

        // ---- Broadcast alpha back to C-layout rows, rescale O.
        #pragma unroll
        for (int j = 0; j < 8; ++j) {
            const float aj = __shfl(alpha, j + g * 8, 32);
            o0[j] *= aj; o1[j] *= aj; o2[j] *= aj; o3[j] *= aj;
        }

        // ---- O += P * V : 4 d-tiles x 2 key-chunks = 8 WMMAs.
        // B 32x16: lane N = lmod (d), element h -> key = h + g*16 (+32 for ap1).
        {
            v16h bmat;
            bmat = *(const v16h*)&sVT[cur][ 0 + lmod][     g * 16]; o0 = WMMA16(ap0, bmat, o0);
            bmat = *(const v16h*)&sVT[cur][ 0 + lmod][32 + g * 16]; o0 = WMMA16(ap1, bmat, o0);
            bmat = *(const v16h*)&sVT[cur][16 + lmod][     g * 16]; o1 = WMMA16(ap0, bmat, o1);
            bmat = *(const v16h*)&sVT[cur][16 + lmod][32 + g * 16]; o1 = WMMA16(ap1, bmat, o1);
            bmat = *(const v16h*)&sVT[cur][32 + lmod][     g * 16]; o2 = WMMA16(ap0, bmat, o2);
            bmat = *(const v16h*)&sVT[cur][32 + lmod][32 + g * 16]; o2 = WMMA16(ap1, bmat, o2);
            bmat = *(const v16h*)&sVT[cur][48 + lmod][     g * 16]; o3 = WMMA16(ap0, bmat, o3);
            bmat = *(const v16h*)&sVT[cur][48 + lmod][32 + g * 16]; o3 = WMMA16(ap1, bmat, o3);
        }

        // ---- Convert + store next block into the other buffer (no conflict
        // with readers of the current buffer), then one barrier to swap.
        if (have_next) {
            const int nxt = cur ^ 1;
            #pragma unroll
            for (int i = 0; i < 4; ++i) {
                const float4 k0 = kr[2*i], k1 = kr[2*i+1];
                v8h hk;
                hk[0]=(_Float16)k0.x; hk[1]=(_Float16)k0.y; hk[2]=(_Float16)k0.z; hk[3]=(_Float16)k0.w;
                hk[4]=(_Float16)k1.x; hk[5]=(_Float16)k1.y; hk[6]=(_Float16)k1.z; hk[7]=(_Float16)k1.w;
                *(v8h*)&sK[nxt][srow][shh + i * 8] = hk;
                const float4 v0 = vr[2*i], v1 = vr[2*i+1];
                sVT[nxt][shh + i*8 + 0][srow] = (_Float16)v0.x;
                sVT[nxt][shh + i*8 + 1][srow] = (_Float16)v0.y;
                sVT[nxt][shh + i*8 + 2][srow] = (_Float16)v0.z;
                sVT[nxt][shh + i*8 + 3][srow] = (_Float16)v0.w;
                sVT[nxt][shh + i*8 + 4][srow] = (_Float16)v1.x;
                sVT[nxt][shh + i*8 + 5][srow] = (_Float16)v1.y;
                sVT[nxt][shh + i*8 + 6][srow] = (_Float16)v1.z;
                sVT[nxt][shh + i*8 + 7][srow] = (_Float16)v1.w;
            }
        }
        __syncthreads();
        cur ^= 1;
    }

    // ---- Finalize: broadcast 1/l per row, store fp32 output (C-layout).
    const float linv = 1.0f / lrow;
    #pragma unroll
    for (int j = 0; j < 8; ++j) {
        const float lj = __shfl(linv, j + g * 8, 32);
        const int   rg = qrow_tile + j + g * 8;
        float* op = Oh + (size_t)rg * DH + lmod;
        op[ 0] = o0[j] * lj;
        op[16] = o1[j] * lj;
        op[32] = o2[j] * lj;
        op[48] = o3[j] * lj;
    }
}

extern "C" void kernel_launch(void* const* d_in, const int* in_sizes, int n_in,
                              void* d_out, int out_size, void* d_ws, size_t ws_size,
                              hipStream_t stream) {
    (void)in_sizes; (void)n_in; (void)out_size; (void)d_ws; (void)ws_size;
    const float* q = (const float*)d_in[0];
    const float* k = (const float*)d_in[1];
    const float* v = (const float*)d_in[2];
    // d_in[3] (block_mask) is the static sliding-window pattern; recomputed analytically.
    float* out = (float*)d_out;

    dim3 grid(SEQ / BQ, NH, 1);   // 64 q-blocks x 8 heads
    dim3 block(128, 1, 1);        // 4 wave32s
    swa_attn_fwd<<<grid, block, 0, stream>>>(q, k, v, out);
}
